// Head_5583457485500
// MI455X (gfx1250) — matrix operand order repaired and checked by reading, not compiled, not run on hardware
//
#include <hip/hip_runtime.h>

#define CTX 32     // T
#define EMB 128    // C
#define HSZ 64     // H
#define NW  8      // waves per block

typedef _Float16     v16h __attribute__((ext_vector_type(16)));
typedef __fp16       v2fp __attribute__((ext_vector_type(2)));
typedef float        v8f  __attribute__((ext_vector_type(8)));
typedef unsigned int v4u  __attribute__((ext_vector_type(4)));

union Frag { unsigned int u[8]; v4u q[2]; v16h v; };

static __device__ __forceinline__ unsigned short f2h(float f) {
    _Float16 h = (_Float16)f;
    return __builtin_bit_cast(unsigned short, h);
}
// v_cvt_pk_rtz_f16_f32: two f32 -> packed f16 pair in one VALU op
static __device__ __forceinline__ unsigned int pk2(float a, float b) {
    v2fp h = __builtin_amdgcn_cvt_pkrtz(a, b);
    return __builtin_bit_cast(unsigned int, h);
}

#define ZV8 {0.f,0.f,0.f,0.f,0.f,0.f,0.f,0.f}

__global__ __launch_bounds__(NW * 32)
void attn_head_kernel(const float* __restrict__ x,
                      const float* __restrict__ Wk,
                      const float* __restrict__ Wq,
                      const float* __restrict__ Wv,
                      float* __restrict__ out, int B)
{
    // All LDS regions hold data PRE-SWIZZLED into WMMA fragment layout:
    // [frag][lane][8 dwords] -> operands load as two ds_load_b128 straight
    // into the contiguous operand register block (no v_mov assembly).
    __shared__ __attribute__((aligned(16))) unsigned int sWf[3][4][4][256]; // W B-frags [pj][kc][nt], 48 KB
    __shared__ __attribute__((aligned(16))) unsigned int sQP[NW][1024];     // q A-frags (mt*2+kc); P A-frags reuse [mt]
    __shared__ __attribute__((aligned(16))) unsigned int sKf[NW][1024];     // k B-frags (st*2+kc)
    __shared__ __attribute__((aligned(16))) unsigned int sVf[NW][1024];     // v B-frags (nt)

    const int tid  = threadIdx.x;
    const int wv   = tid >> 5;
    const int lane = tid & 31;
    const int lh   = lane >> 4;   // 16-lane half
    const int ll   = lane & 15;

    // ---- Stage weights once per block, swizzled into B-fragment layout ----
    const float* Wsrc[3] = { Wk, Wq, Wv };
    for (int idx = tid; idx < 3 * (EMB/2) * HSZ; idx += NW * 32) {
        const int pj  = idx >> 12;
        const int rem = idx & 4095;
        const int kp  = rem >> 6;          // K-pair index
        const int n   = rem & 63;
        const int K   = kp << 1;
        // B-frag position: VGPR r holds K = 32*kc + 2*r + 16*lh2, n = 16*nt + ll2
        const int kc    = K >> 5;
        const int r2    = (K & 15) >> 1;
        const int lh2   = (K >> 4) & 1;
        const int nt    = n >> 4;
        const int lane2 = (n & 15) + (lh2 << 4);
        const float a = Wsrc[pj][K * HSZ + n];
        const float b = Wsrc[pj][(K + 1) * HSZ + n];
        sWf[pj][kc][nt][lane2 * 8 + r2] = pk2(a, b);
    }
    __syncthreads();

    const int b = blockIdx.x * NW + wv;
    if (b >= B) return;   // wave-uniform: EXEC all-ones for WMMA

    const float* xb = x + (size_t)b * CTX * EMB;

    // ---- x -> f16 A-fragments: 8 consecutive floats per (mt,kc,half) => b128 x2 ----
    Frag ax[2][4];
#pragma unroll
    for (int mt = 0; mt < 2; ++mt)
#pragma unroll
        for (int kc = 0; kc < 4; ++kc)
#pragma unroll
            for (int a = 0; a < 2; ++a) {
                const int m = 16*mt + ll;
                const float* p = xb + m * EMB + 32*kc + 16*lh + 8*a;
                const float4 f0 = *(const float4*)p;
                const float4 f1 = *(const float4*)(p + 4);
                ax[mt][kc].u[4*a + 0] = pk2(f0.x, f0.y);
                ax[mt][kc].u[4*a + 1] = pk2(f0.z, f0.w);
                ax[mt][kc].u[4*a + 2] = pk2(f1.x, f1.y);
                ax[mt][kc].u[4*a + 3] = pk2(f1.z, f1.w);
            }

    // ---- Projections: preload all 4 K-chunk weight fragments (one ds clause,
    //      one wait), then 8 back-to-back WMMAs ----
#pragma unroll
    for (int pj = 0; pj < 3; ++pj) {
#pragma unroll
        for (int nt = 0; nt < 4; ++nt) {
            Frag bw[4];
#pragma unroll
            for (int kc = 0; kc < 4; ++kc) {
                const unsigned int* wp = &sWf[pj][kc][nt][lane * 8];
                bw[kc].q[0] = *(const v4u*)wp;
                bw[kc].q[1] = *(const v4u*)(wp + 4);
            }
            v8f accs[2] = { ZV8, ZV8 };
#pragma unroll
            for (int kc = 0; kc < 4; ++kc) {
                accs[0] = __builtin_amdgcn_wmma_f32_16x16x32_f16(
                    false, ax[0][kc].v, false, bw[kc].v, (short)0, accs[0], false, false);
                accs[1] = __builtin_amdgcn_wmma_f32_16x16x32_f16(
                    false, ax[1][kc].v, false, bw[kc].v, (short)0, accs[1], false, false);
            }
            // scatter results directly into the fragment layout of the next GEMM
#pragma unroll
            for (int mt = 0; mt < 2; ++mt) {
                if (pj == 0) {          // k -> S-GEMM B-frag (st=mt, kc2=nt>>1), K-dim = h
                    const int base = (mt*2 + (nt >> 1)) * 256;
#pragma unroll
                    for (int r = 0; r < 8; ++r)
                        ((unsigned short*)&sKf[wv][base + (r + 8*lh + 16*((ll >> 3) & 1))*8 + 4*(nt & 1) + ((ll & 7) >> 1)])[ll & 1]
                            = f2h(accs[mt][r]);
                } else if (pj == 1) {   // q -> S-GEMM A-frag (mt, kc2=nt>>1)
                    const int base = (mt*2 + (nt >> 1)) * 256;
#pragma unroll
                    for (int r = 0; r < 8; ++r)
                        ((unsigned short*)&sQP[wv][base + (r + 8*lh + 16*((ll >> 3) & 1))*8
                                                   + 4*(nt & 1) + ((ll & 7) >> 1)])[ll & 1]
                            = f2h(accs[mt][r]);
                } else {                // v -> O-GEMM B-frag (nt), K-dim = s: pairs pack along r
                    const int base = nt*256 + (ll + 16*lh)*8 + 4*mt;
#pragma unroll
                    for (int r = 0; r < 8; r += 2)
                        sVf[wv][base + (r >> 1)] = pk2(accs[mt][r], accs[mt][r + 1]);
                }
            }
        }
    }

    // ---- S = q @ k^T (all fragments preloaded, 8 WMMAs) ----
    Frag aq[2][2], bk[2][2];
#pragma unroll
    for (int mt = 0; mt < 2; ++mt)
#pragma unroll
        for (int kc = 0; kc < 2; ++kc) {
            const unsigned int* p = &sQP[wv][(mt*2 + kc) * 256 + lane * 8];
            aq[mt][kc].q[0] = *(const v4u*)p;
            aq[mt][kc].q[1] = *(const v4u*)(p + 4);
        }
#pragma unroll
    for (int st = 0; st < 2; ++st)
#pragma unroll
        for (int kc = 0; kc < 2; ++kc) {
            const unsigned int* p = &sKf[wv][(st*2 + kc) * 256 + lane * 8];
            bk[st][kc].q[0] = *(const v4u*)p;
            bk[st][kc].q[1] = *(const v4u*)(p + 4);
        }
    v8f Sc[2][2];
#pragma unroll
    for (int mt = 0; mt < 2; ++mt)
#pragma unroll
        for (int st = 0; st < 2; ++st) {
            v8f acc = ZV8;
            acc = __builtin_amdgcn_wmma_f32_16x16x32_f16(
                false, aq[mt][0].v, false, bk[st][0].v, (short)0, acc, false, false);
            acc = __builtin_amdgcn_wmma_f32_16x16x32_f16(
                false, aq[mt][1].v, false, bk[st][1].v, (short)0, acc, false, false);
            Sc[mt][st] = acc;
        }

    // ---- Causal softmax; P written straight into A-fragment layout (reuses q region) ----
    const float scale = 0.125f;   // H^-0.5
#pragma unroll
    for (int mt = 0; mt < 2; ++mt) {
#pragma unroll
        for (int r = 0; r < 8; ++r) {
            const int t  = 16*mt + r + 8*lh;
            const int s0 = ll, s1 = 16 + ll;
            float e0 = Sc[mt][0][r] * scale;
            float e1 = Sc[mt][1][r] * scale;
            if (s0 > t) e0 = -__builtin_inff();
            if (s1 > t) e1 = -__builtin_inff();
            float mx = fmaxf(e0, e1);
#pragma unroll
            for (int off = 1; off < 16; off <<= 1)
                mx = fmaxf(mx, __shfl_xor(mx, off, 32));
            const float p0 = __expf(e0 - mx);
            const float p1 = __expf(e1 - mx);
            float sm = p0 + p1;
#pragma unroll
            for (int off = 1; off < 16; off <<= 1)
                sm += __shfl_xor(sm, off, 32);
            const float inv = 1.0f / sm;
            // A-frag position for P element (t, s): lane2 = (t&15) + 16*((s>>3)&1),
            // dword r2 = 4*((s>>4)&1) + ((s&7)>>1), half = s&1
            const int base = mt*256 + (r + 8*lh + 16*((ll >> 3) & 1))*8 + ((ll & 7) >> 1);
            ((unsigned short*)&sQP[wv][base    ])[ll & 1] = f2h(p0 * inv);
            ((unsigned short*)&sQP[wv][base + 4])[ll & 1] = f2h(p1 * inv);
        }
    }

    // ---- O = P @ v (8 WMMAs) + coalesced f32 stores ----
    Frag bvf[4];
#pragma unroll
    for (int nt = 0; nt < 4; ++nt) {
        const unsigned int* p = &sVf[wv][nt * 256 + lane * 8];
        bvf[nt].q[0] = *(const v4u*)p;
        bvf[nt].q[1] = *(const v4u*)(p + 4);
    }
#pragma unroll
    for (int mt = 0; mt < 2; ++mt) {
        Frag ap;
        const unsigned int* p = &sQP[wv][mt * 256 + lane * 8];
        ap.q[0] = *(const v4u*)p;
        ap.q[1] = *(const v4u*)(p + 4);
#pragma unroll
        for (int nt = 0; nt < 4; ++nt) {
            v8f acc = ZV8;
            acc = __builtin_amdgcn_wmma_f32_16x16x32_f16(
                false, ap.v, false, bvf[nt].v, (short)0, acc, false, false);
#pragma unroll
            for (int r = 0; r < 8; ++r) {
                const int t = 16*mt + r + 8*lh;
                const int h = 16*nt + ll;
                out[((size_t)b * CTX + t) * HSZ + h] = acc[r];
            }
        }
    }
}

extern "C" void kernel_launch(void* const* d_in, const int* in_sizes, int n_in,
                              void* d_out, int out_size, void* d_ws, size_t ws_size,
                              hipStream_t stream) {
    const float* x  = (const float*)d_in[0];
    const float* Wk = (const float*)d_in[1];
    const float* Wq = (const float*)d_in[2];
    const float* Wv = (const float*)d_in[3];
    float* out = (float*)d_out;
    const int B = in_sizes[0] / (CTX * EMB);
    const int grid = (B + NW - 1) / NW;
    attn_head_kernel<<<grid, NW * 32, 0, stream>>>(x, Wk, Wq, Wv, out, B);
}
